// W8A8B8O8LinearParallel_40243843564268
// MI455X (gfx1250) — 8-bit matrix operand order repaired and checked by reading, compiled, not run on hardware
//
#include <hip/hip_runtime.h>
#include <stdint.h>

typedef __attribute__((ext_vector_type(8))) int v8i;
typedef __attribute__((ext_vector_type(4))) int v4i;
typedef __attribute__((ext_vector_type(2))) int v2i;

#define K_DIM 4096
#define N_DIM 4096
#define BLK_M 128
#define BLK_N 256
#define BLK_K 64
#define LDA 80
#define LDB 80
#define KSTEPS (K_DIM / BLK_K)

__device__ __forceinline__ unsigned perm_b32(unsigned hi, unsigned lo, unsigned sel) {
  return __builtin_amdgcn_perm(hi, lo, sel);
}

__device__ __forceinline__ unsigned pack4(unsigned a, unsigned b, unsigned c, unsigned d) {
  unsigned t0 = perm_b32(b, a, 0x00000400u);
  unsigned t1 = perm_b32(d, c, 0x00000400u);
  return perm_b32(t1, t0, 0x05040100u);
}


__device__ __forceinline__ void frag_compute(const int8_t* __restrict__ ab,
                                             const int8_t* __restrict__ bb,
                                             int wm, int wn, int lane,
                                             v8i acc[4][4]) {
  const int lr = lane & 15;
  const int ko = (lane >> 4) * 16;
  const int kh = (lane >> 4) * 16;

  v8i af[4];
#pragma unroll
  for (int mt = 0; mt < 4; ++mt) {
    const int8_t* p = ab + (wm + mt * 16 + lr) * LDA + ko;
    v2i x0 = *(const v2i*)(p);
    v2i x1 = *(const v2i*)(p + 8);
    v2i x2 = *(const v2i*)(p + 32);
    v2i x3 = *(const v2i*)(p + 40);
    af[mt] = (v8i){x0.x, x0.y, x1.x, x1.y, x2.x, x2.y, x3.x, x3.y};
  }
  v8i bf[4];
#pragma unroll
  for (int nt = 0; nt < 4; ++nt) {
    const int8_t* p = bb + (wn + nt * 16 + lr) * LDB + kh;
    v4i y0 = *(const v4i*)(p);
    v4i y1 = *(const v4i*)(p + 32);
    bf[nt] = (v8i){y0.x, y0.y, y0.z, y0.w, y1.x, y1.y, y1.z, y1.w};
  }
#pragma unroll
  for (int mt = 0; mt < 4; ++mt)
#pragma unroll
    for (int nt = 0; nt < 4; ++nt)
      acc[mt][nt] = __builtin_amdgcn_wmma_i32_16x16x64_iu8(
           true, af[mt],  true, bf[nt],
          acc[mt][nt],  false,  false);
}

__device__ __forceinline__ void epilogue(v8i acc[4][4], const float* __restrict__ a_s,
                                         const float* __restrict__ b_s,
                                         float* __restrict__ Out,
                                         int mblk, int nblk, int wm, int wn, int lane) {
  const float a0 = a_s[0];
  const int lr = lane & 15;
  const int lh = (lane >> 4) * 8;
#pragma unroll
  for (int nt = 0; nt < 4; ++nt) {
    const int col = nblk + wn + nt * 16 + lr;
    const float sb = a0 * b_s[col];
#pragma unroll
    for (int mt = 0; mt < 4; ++mt) {
#pragma unroll
      for (int v = 0; v < 8; ++v) {
        const int row = mblk + wm + mt * 16 + lh + v;
        float f = (float)acc[mt][nt][v] * sb;
        f = fminf(fmaxf(__builtin_rintf(f), -128.0f), 127.0f);
        Out[(size_t)row * N_DIM + col] = f;
      }
    }
  }
}


__global__ __launch_bounds__(256)
void pack_x_kernel(const int* __restrict__ X, int8_t* __restrict__ XP, int n4) {
  const int idx = blockIdx.x * 256 + threadIdx.x;
  if (idx < n4) {
    v4i v = ((const v4i*)X)[idx];
    ((unsigned*)XP)[idx] =
        pack4((unsigned)v.x, (unsigned)v.y, (unsigned)v.z, (unsigned)v.w);
  }
}

__global__ __launch_bounds__(256)
void pack_w_kernel(const int* __restrict__ W, int8_t* __restrict__ WT) {
  __shared__ __align__(16) int8_t tb[64 * 64];
  const int tid = threadIdx.x;
  const int kb = blockIdx.x * 64;
  const int nb = blockIdx.y * 64;
  const int cg = (tid & 15) * 4;
  const int kg = (tid >> 4) * 4;

  const int* wp = W + (size_t)(kb + kg) * N_DIM + nb + cg;
  v4i r0 = *(const v4i*)(wp);
  v4i r1 = *(const v4i*)(wp + N_DIM);
  v4i r2 = *(const v4i*)(wp + 2 * N_DIM);
  v4i r3 = *(const v4i*)(wp + 3 * N_DIM);
  *(unsigned*)(tb + (cg + 0) * 64 + kg) =
      pack4((unsigned)r0.x, (unsigned)r1.x, (unsigned)r2.x, (unsigned)r3.x);
  *(unsigned*)(tb + (cg + 1) * 64 + kg) =
      pack4((unsigned)r0.y, (unsigned)r1.y, (unsigned)r2.y, (unsigned)r3.y);
  *(unsigned*)(tb + (cg + 2) * 64 + kg) =
      pack4((unsigned)r0.z, (unsigned)r1.z, (unsigned)r2.z, (unsigned)r3.z);
  *(unsigned*)(tb + (cg + 3) * 64 + kg) =
      pack4((unsigned)r0.w, (unsigned)r1.w, (unsigned)r2.w, (unsigned)r3.w);
  __syncthreads();

  const int col = tid >> 2;
  const int seg = (tid & 3) * 16;
  v4i v = *(const v4i*)(tb + col * 64 + seg);
  *(v4i*)(WT + (size_t)(nb + col) * K_DIM + kb + seg) = v;
}


__global__ __launch_bounds__(256)
void w8a8_gemm_pre(const int8_t* __restrict__ XP, const int8_t* __restrict__ WT,
                   const float* __restrict__ a_s, const float* __restrict__ b_s,
                   float* __restrict__ Out) {
  __shared__ __align__(16) int8_t lA[2][BLK_M * LDA];
  __shared__ __align__(16) int8_t lB[2][BLK_N * LDB];

  const int tid  = threadIdx.x;
  const int lane = tid & 31;
  const int wave = tid >> 5;
  const int wm   = (wave >> 2) * 64;
  const int wn   = (wave & 3) * 64;
  const int mblk = blockIdx.y * BLK_M;
  const int nblk = blockIdx.x * BLK_N;

  const int arow = tid >> 1;
  const int aoff = (tid & 1) * 32;

  v4i ra0, ra1, rb[4];

  auto g_load = [&](int ks) {
    const int8_t* xp = XP + (size_t)(mblk + arow) * K_DIM + ks + aoff;
    ra0 = *(const v4i*)(xp);
    ra1 = *(const v4i*)(xp + 16);
    const int8_t* wp = WT + (size_t)(nblk + tid) * K_DIM + ks;
#pragma unroll
    for (int j = 0; j < 4; ++j) rb[j] = *(const v4i*)(wp + j * 16);
  };

  auto l_store = [&](int buf) {
    int8_t* ab = &lA[buf][0];
    *(v4i*)(ab + arow * LDA + aoff) = ra0;
    *(v4i*)(ab + arow * LDA + aoff + 16) = ra1;
    int8_t* bb = &lB[buf][0];
#pragma unroll
    for (int j = 0; j < 4; ++j) *(v4i*)(bb + tid * LDB + j * 16) = rb[j];
  };

  v8i acc[4][4];
#pragma unroll
  for (int mt = 0; mt < 4; ++mt)
#pragma unroll
    for (int nt = 0; nt < 4; ++nt)
      acc[mt][nt] = (v8i){0, 0, 0, 0, 0, 0, 0, 0};

  g_load(0);
  l_store(0);
  __syncthreads();

  for (int kt = 0; kt < KSTEPS; ++kt) {
    const int cur = kt & 1;
    if (kt + 1 < KSTEPS) g_load((kt + 1) * BLK_K);
    frag_compute(&lA[cur][0], &lB[cur][0], wm, wn, lane, acc);
    if (kt + 1 < KSTEPS) l_store(cur ^ 1);
    __syncthreads();
  }

  epilogue(acc, a_s, b_s, Out, mblk, nblk, wm, wn, lane);
}


__global__ __launch_bounds__(256)
void w8a8_gemm_fused(const int* __restrict__ X, const int* __restrict__ W,
                     const float* __restrict__ a_s, const float* __restrict__ b_s,
                     float* __restrict__ Out) {
  __shared__ __align__(16) int8_t lA[2][BLK_M * LDA];
  __shared__ __align__(16) int8_t lB[2][BLK_N * LDB];

  const int tid  = threadIdx.x;
  const int lane = tid & 31;
  const int wave = tid >> 5;
  const int wm   = (wave >> 2) * 64;
  const int wn   = (wave & 3) * 64;
  const int mblk = blockIdx.y * BLK_M;
  const int nblk = blockIdx.x * BLK_N;

  const int arow = tid >> 1;
  const int aoff = (tid & 1) * 32;
  const int bcg  = (tid & 63) * 4;
  const int bkg  = (tid >> 6) * 4;

  unsigned pa[8];
  unsigned pw[4][4];

  auto g_load = [&](int ks) {
    const int* xp = X + (size_t)(mblk + arow) * K_DIM + ks + aoff;
#pragma unroll
    for (int i = 0; i < 8; ++i) {
      v4i x = *(const v4i*)(xp + i * 4);
      pa[i] = pack4((unsigned)x.x, (unsigned)x.y, (unsigned)x.z, (unsigned)x.w);
    }
#pragma unroll
    for (int u = 0; u < 4; ++u) {
      const int kg = bkg + u * 16;
      const int* wp = W + (size_t)(ks + kg) * N_DIM + nblk + bcg;
      v4i r0 = *(const v4i*)(wp);
      v4i r1 = *(const v4i*)(wp + N_DIM);
      v4i r2 = *(const v4i*)(wp + 2 * N_DIM);
      v4i r3 = *(const v4i*)(wp + 3 * N_DIM);
      pw[u][0] = pack4((unsigned)r0.x, (unsigned)r1.x, (unsigned)r2.x, (unsigned)r3.x);
      pw[u][1] = pack4((unsigned)r0.y, (unsigned)r1.y, (unsigned)r2.y, (unsigned)r3.y);
      pw[u][2] = pack4((unsigned)r0.z, (unsigned)r1.z, (unsigned)r2.z, (unsigned)r3.z);
      pw[u][3] = pack4((unsigned)r0.w, (unsigned)r1.w, (unsigned)r2.w, (unsigned)r3.w);
    }
  };

  auto l_store = [&](int buf) {
    int8_t* ab = &lA[buf][0];
    *(v4i*)(ab + arow * LDA + aoff) =
        (v4i){(int)pa[0], (int)pa[1], (int)pa[2], (int)pa[3]};
    *(v4i*)(ab + arow * LDA + aoff + 16) =
        (v4i){(int)pa[4], (int)pa[5], (int)pa[6], (int)pa[7]};
    int8_t* bb = &lB[buf][0];
#pragma unroll
    for (int u = 0; u < 4; ++u) {
      const int kg = bkg + u * 16;
#pragma unroll
      for (int j = 0; j < 4; ++j)
        *(unsigned*)(bb + (bcg + j) * LDB + kg) = pw[u][j];
    }
  };

  v8i acc[4][4];
#pragma unroll
  for (int mt = 0; mt < 4; ++mt)
#pragma unroll
    for (int nt = 0; nt < 4; ++nt)
      acc[mt][nt] = (v8i){0, 0, 0, 0, 0, 0, 0, 0};

  g_load(0);
  l_store(0);
  __syncthreads();

  for (int kt = 0; kt < KSTEPS; ++kt) {
    const int cur = kt & 1;
    if (kt + 1 < KSTEPS) g_load((kt + 1) * BLK_K);
    frag_compute(&lA[cur][0], &lB[cur][0], wm, wn, lane, acc);
    if (kt + 1 < KSTEPS) l_store(cur ^ 1);
    __syncthreads();
  }

  epilogue(acc, a_s, b_s, Out, mblk, nblk, wm, wn, lane);
}

extern "C" void kernel_launch(void* const* d_in, const int* in_sizes, int n_in,
                              void* d_out, int out_size, void* d_ws, size_t ws_size,
                              hipStream_t stream) {
  const int*   X = (const int*)d_in[0];
  const int*   W = (const int*)d_in[1];
  const float* a = (const float*)d_in[2];
  const float* b = (const float*)d_in[3];
  float* out = (float*)d_out;
  const int M = in_sizes[0] / K_DIM;

  const size_t pack_bytes = (size_t)M * K_DIM + (size_t)K_DIM * N_DIM;
  dim3 grid(N_DIM / BLK_N, M / BLK_M);

  if (ws_size >= pack_bytes) {
    int8_t* xp = (int8_t*)d_ws;
    int8_t* wt = xp + (size_t)M * K_DIM;
    const int n4 = (M * K_DIM) / 4;
    pack_x_kernel<<<(n4 + 255) / 256, 256, 0, stream>>>(X, xp, n4);
    pack_w_kernel<<<dim3(K_DIM / 64, N_DIM / 64), 256, 0, stream>>>(W, wt);
    w8a8_gemm_pre<<<grid, dim3(256), 0, stream>>>(xp, wt, a, b, out);
  } else {
    w8a8_gemm_fused<<<grid, dim3(256), 0, stream>>>(X, W, a, b, out);
  }
}
